// DotProductAttention_19971597926954
// MI455X (gfx1250) — matrix operand order repaired and checked by reading, not compiled, not run on hardware
//
#include <hip/hip_runtime.h>
#include <hip/hip_bf16.h>
#include <stdint.h>

// out = rope(x @ rope_w + rope_b) @ proj_w + proj_b  (rest of reference is dead code)
//
// Pipeline:
//   prep:  x -> f16 xh;  rope_w/proj_w -> f16 transposed [n][k]
//   gemm1: xr = rope(xh @ rwT^T + rope_b)  -> f16 workspace
//   gemm2: out = xr @ pwT^T + proj_b       -> f32 d_out
// GEMM main loops are double-buffered with async global->LDS loads (ASYNCcnt).

typedef __attribute__((ext_vector_type(16))) _Float16 v16h;
typedef __attribute__((ext_vector_type(8)))  _Float16 v8h;
typedef __attribute__((ext_vector_type(4)))  _Float16 v4h;
typedef __attribute__((ext_vector_type(2)))  _Float16 v2h;
typedef __attribute__((ext_vector_type(8)))  float    v8f;
typedef __attribute__((ext_vector_type(4)))  float    v4f;
typedef __attribute__((ext_vector_type(4)))  int      v4i;

#define DIM    1024
#define MTOT   16384       // B*N rows
#define NTOK   4096
#define BT     128         // M/N tile per block
#define KT     64          // K per LDS stage (2 wmma k-steps)
#define AP     72          // LDS pitch in halfs (row = 144B, 16B aligned)
#define ASZ    (128 * AP * 2)   // bytes per A or B stage  (18432)
#define STAGE  (2 * ASZ)        // bytes per double-buffer slot (36864)
#define CPITCH 128

// ---- async global->LDS (CDNA5) with safe fallback --------------------------
__device__ __forceinline__ void async_copy16(const _Float16* g, _Float16* l) {
#if __has_builtin(__builtin_amdgcn_global_load_async_to_lds_b128)
    __builtin_amdgcn_global_load_async_to_lds_b128(
        (__attribute__((address_space(1))) v4i*)g,
        (__attribute__((address_space(3))) v4i*)l, 0, 0);
#else
    *(v8h*)l = *(const v8h*)g;
#endif
}

__device__ __forceinline__ void wait_async0() {
#if __has_builtin(__builtin_amdgcn_global_load_async_to_lds_b128)
#if __has_builtin(__builtin_amdgcn_s_wait_asynccnt)
    __builtin_amdgcn_s_wait_asynccnt(0);
#else
    asm volatile("s_wait_asynccnt 0x0" ::: "memory");
#endif
#endif
}

// ---- prep kernels ----------------------------------------------------------
__global__ __launch_bounds__(256) void k_cvt_f16(const float* __restrict__ src,
                                                 _Float16* __restrict__ dst) {
    size_t id = (size_t)blockIdx.x * 256 + threadIdx.x;   // 4 floats each
    v4f v = *(const v4f*)(src + id * 4);
    v4h h; h[0] = (_Float16)v[0]; h[1] = (_Float16)v[1];
           h[2] = (_Float16)v[2]; h[3] = (_Float16)v[3];
    *(v4h*)(dst + id * 4) = h;
}

__global__ __launch_bounds__(256) void k_cvt_transpose(const float* __restrict__ w,
                                                       _Float16* __restrict__ wt) {
    size_t id = (size_t)blockIdx.x * 256 + threadIdx.x;   // 1M elements
    int n = (int)(id & (DIM - 1));
    int k = (int)(id >> 10);
    wt[(size_t)n * DIM + k] = (_Float16)w[(size_t)k * DIM + n];  // wt: [n][k]
}

// ---- shared GEMM pieces ----------------------------------------------------
__device__ __forceinline__ void load_stage(const _Float16* __restrict__ A,
                                           const _Float16* __restrict__ Bt,
                                           _Float16* As, _Float16* Bs,
                                           int m0, int n0, int kk, int tid) {
    #pragma unroll
    for (int it = 0; it < 4; ++it) {
        int id  = tid + it * 256;       // 1024 chunks of 8 halfs per operand
        int row = id >> 3;
        int c8  = (id & 7) << 3;
        async_copy16(A  + (size_t)(m0 + row) * DIM + kk + c8, As + row * AP + c8);
        async_copy16(Bt + (size_t)(n0 + row) * DIM + kk + c8, Bs + row * AP + c8);
    }
}

__device__ __forceinline__ void compute_stage(const _Float16* As, const _Float16* Bs,
                                              int wrow, int wcol, int lane,
                                              v8f (&acc)[2][4]) {
    #pragma unroll
    for (int ks = 0; ks < KT; ks += 32) {
        v16h a[2], b[4];
        #pragma unroll
        for (int tm = 0; tm < 2; ++tm) {    // A 16x32: lane half selects K octet
            const _Float16* p = As + (wrow + tm * 16 + (lane & 15)) * AP
                                   + ks + ((lane >> 4) << 3);
            v8h lo = *(const v8h*)p;
            v8h hi = *(const v8h*)(p + 16);
            #pragma unroll
            for (int i = 0; i < 8; ++i) { a[tm][i] = lo[i]; a[tm][i + 8] = hi[i]; }
        }
        #pragma unroll
        for (int tn = 0; tn < 4; ++tn) {    // B 32x16: lane half selects K octet
            const _Float16* p = Bs + (wcol + tn * 16 + (lane & 15)) * AP
                                   + ks + ((lane >> 4) << 3);
            v8h lo = *(const v8h*)p;
            v8h hi = *(const v8h*)(p + 16);
            #pragma unroll
            for (int i = 0; i < 8; ++i) { b[tn][i] = lo[i]; b[tn][i + 8] = hi[i]; }
        }
        #pragma unroll
        for (int tm = 0; tm < 2; ++tm)
            #pragma unroll
            for (int tn = 0; tn < 4; ++tn)
                acc[tm][tn] = __builtin_amdgcn_wmma_f32_16x16x32_f16(
                    false, a[tm], false, b[tn], (short)0, acc[tm][tn], false, false);
    }
}

// ---------------------------------------------------------------------------
// Kernel 1: xr = rope(xh @ rwT^T + rope_b) -> f16 workspace
// ---------------------------------------------------------------------------
__global__ __launch_bounds__(256) void k_gemm_rope(
    const _Float16* __restrict__ xh, const _Float16* __restrict__ rwT,
    const float* __restrict__ rope_b, const float* __restrict__ freqs,
    _Float16* __restrict__ xr)
{
    __shared__ __align__(16) char smem[2 * STAGE];   // 73728 B; epilogue aliases
    const int tid  = threadIdx.x;
    const int lane = tid & 31;
    const int wave = tid >> 5;
    const int wrow = (wave & 3) * 32;
    const int wcol = (wave >> 2) * 64;
    const int m0 = blockIdx.y * BT;
    const int n0 = blockIdx.x * BT;

    v8f acc[2][4];
    #pragma unroll
    for (int i = 0; i < 2; ++i)
        #pragma unroll
        for (int j = 0; j < 4; ++j) acc[i][j] = (v8f){};

    load_stage(xh, rwT, (_Float16*)smem, (_Float16*)(smem + ASZ), m0, n0, 0, tid);
    wait_async0();
    __syncthreads();

    for (int kk = 0; kk < DIM; kk += KT) {
        int cur = (kk >> 6) & 1;
        char* cb = smem + cur * STAGE;
        char* nb = smem + (cur ^ 1) * STAGE;
        if (kk + KT < DIM)
            load_stage(xh, rwT, (_Float16*)nb, (_Float16*)(nb + ASZ),
                       m0, n0, kk + KT, tid);
        compute_stage((const _Float16*)cb, (const _Float16*)(cb + ASZ),
                      wrow, wcol, lane, acc);
        wait_async0();
        __syncthreads();
    }

    // ---- epilogue: +rope_b, stage f32 tile, apply RoPE, store f16 xr ----
    float* Cs = (float*)smem;                       // [128][128], aliases stages
    #pragma unroll
    for (int tm = 0; tm < 2; ++tm)
        #pragma unroll
        for (int tn = 0; tn < 4; ++tn) {
            int r0 = wrow + tm * 16 + ((lane >> 4) << 3);
            int c  = wcol + tn * 16 + (lane & 15);
            float bias = rope_b[n0 + c];
            #pragma unroll
            for (int v = 0; v < 8; ++v)
                Cs[(r0 + v) * CPITCH + c] = acc[tm][tn][v] + bias;
        }
    __syncthreads();

    #pragma unroll 4
    for (int it = 0; it < 32; ++it) {
        int id  = tid + it * 256;      // 8192 (x1,x2) pairs in tile
        int row = id >> 6;
        int col = (id & 63) << 1;
        float x1 = Cs[row * CPITCH + col];
        float x2 = Cs[row * CPITCH + col + 1];
        int tok = (m0 + row) & (NTOK - 1);
        int j   = (col & 63) >> 1;     // n0 is a multiple of 64
        float f  = freqs[tok * 32 + j];
        float cs = __cosf(f), sn = __sinf(f);
        v2h o;
        o[0] = (_Float16)(x1 * cs - x2 * sn);
        o[1] = (_Float16)(x1 * sn + x2 * cs);
        *(v2h*)(xr + (size_t)(m0 + row) * DIM + n0 + col) = o;
    }
}

// ---------------------------------------------------------------------------
// Kernel 2: out = xr @ pwT^T + proj_b (f32 output)
// ---------------------------------------------------------------------------
__global__ __launch_bounds__(256) void k_gemm_proj(
    const _Float16* __restrict__ xr, const _Float16* __restrict__ pwT,
    const float* __restrict__ proj_b, float* __restrict__ out)
{
    __shared__ __align__(16) char smem[2 * STAGE];
    const int tid  = threadIdx.x;
    const int lane = tid & 31;
    const int wave = tid >> 5;
    const int wrow = (wave & 3) * 32;
    const int wcol = (wave >> 2) * 64;
    const int m0 = blockIdx.y * BT;
    const int n0 = blockIdx.x * BT;

    v8f acc[2][4];
    #pragma unroll
    for (int i = 0; i < 2; ++i)
        #pragma unroll
        for (int j = 0; j < 4; ++j) acc[i][j] = (v8f){};

    load_stage(xr, pwT, (_Float16*)smem, (_Float16*)(smem + ASZ), m0, n0, 0, tid);
    wait_async0();
    __syncthreads();

    for (int kk = 0; kk < DIM; kk += KT) {
        int cur = (kk >> 6) & 1;
        char* cb = smem + cur * STAGE;
        char* nb = smem + (cur ^ 1) * STAGE;
        if (kk + KT < DIM)
            load_stage(xr, pwT, (_Float16*)nb, (_Float16*)(nb + ASZ),
                       m0, n0, kk + KT, tid);
        compute_stage((const _Float16*)cb, (const _Float16*)(cb + ASZ),
                      wrow, wcol, lane, acc);
        wait_async0();
        __syncthreads();
    }

    #pragma unroll
    for (int tm = 0; tm < 2; ++tm)
        #pragma unroll
        for (int tn = 0; tn < 4; ++tn) {
            int r0 = wrow + tm * 16 + ((lane >> 4) << 3);
            int c  = n0 + wcol + tn * 16 + (lane & 15);
            float bias = proj_b[c];
            #pragma unroll
            for (int v = 0; v < 8; ++v)
                out[(size_t)(m0 + r0 + v) * DIM + c] = acc[tm][tn][v] + bias;
        }
}

extern "C" void kernel_launch(void* const* d_in, const int* in_sizes, int n_in,
                              void* d_out, int out_size, void* d_ws, size_t ws_size,
                              hipStream_t stream) {
    (void)in_sizes; (void)n_in; (void)out_size; (void)ws_size;
    const float* x      = (const float*)d_in[0];
    const float* freqs  = (const float*)d_in[1];
    const float* rope_w = (const float*)d_in[2];
    const float* rope_b = (const float*)d_in[3];
    const float* proj_w = (const float*)d_in[15];
    const float* proj_b = (const float*)d_in[16];

    _Float16* xh  = (_Float16*)d_ws;                  // 32 MB
    _Float16* rwT = xh  + (size_t)MTOT * DIM;         //  2 MB
    _Float16* pwT = rwT + (size_t)DIM * DIM;          //  2 MB
    _Float16* xr  = pwT + (size_t)DIM * DIM;          // 32 MB

    k_cvt_f16<<<(MTOT * DIM) / (256 * 4), 256, 0, stream>>>(x, xh);
    k_cvt_transpose<<<(DIM * DIM) / 256, 256, 0, stream>>>(rope_w, rwT);
    k_cvt_transpose<<<(DIM * DIM) / 256, 256, 0, stream>>>(proj_w, pwT);

    dim3 grid(DIM / BT, MTOT / BT);    // (8, 128)
    k_gemm_rope<<<grid, 256, 0, stream>>>(xh, rwT, rope_b, freqs, xr);
    k_gemm_proj<<<grid, 256, 0, stream>>>(xr, pwT, proj_b, (float*)d_out);
}
